// MyGAT_86002425135606
// MI455X (gfx1250) — matrix operand order repaired and checked by reading, not compiled, not run on hardware
//
#include <hip/hip_runtime.h>
#include <hip/hip_bf16.h>
#include <math.h>

typedef __attribute__((ext_vector_type(16))) _Float16 v16h;
typedef __attribute__((ext_vector_type(8)))  float    v8f;

// ---------------------------------------------------------------------------
// Float atomic max via monotone int/uint encoding (init value must be -inf).
// ---------------------------------------------------------------------------
__device__ __forceinline__ void atomicMaxF(float* addr, float val) {
  if (val >= 0.0f)
    atomicMax((int*)addr, __float_as_int(val));
  else
    atomicMin((unsigned int*)addr, __float_as_uint(val));
}

// ---------------------------------------------------------------------------
// Pre-swizzle weight matrix B[K,Ncols] (f32, row-major) into the f16 WMMA
// B-operand layout: Bsw[((kt*2+gi)*Ncols + n)*16 + j] = B[(kt*32+gi*8+(j&7)+(j>>3)*16)*Ncols+n]
// so each lane's 16 B-halfs for one k-step are a contiguous 32-byte chunk.
// ---------------------------------------------------------------------------
__global__ void swizzle_w_f16(const float* __restrict__ B,
                              _Float16* __restrict__ Bsw, int K, int Ncols) {
  const int t = blockIdx.x * blockDim.x + threadIdx.x;
  if (t >= K * Ncols) return;
  const int j = t & 15;
  const int rest = t >> 4;
  const int n = rest % Ncols;
  const int kg = rest / Ncols;            // kt*2 + gi
  const int k = (kg >> 1) * 32 + (kg & 1) * 8 + (j & 7) + (j >> 3) * 16;
  Bsw[t] = (_Float16)B[(size_t)k * Ncols + n];
}

// ---------------------------------------------------------------------------
// WMMA GEMM:  C[M,Ncols] = A[M,128] * B[128,Ncols], f32 A/C, f16 compute.
// One wave32 computes CT adjacent 16x16 tiles (A registers reused across CT).
// A layout (ISA 7.12.2, 16-bit A 16x32): lane<16 -> K=0..7,16..23;
// lane>=16 -> K=8..15,24..31; row M = lane&15.
// B operand read pre-swizzled (contiguous v16h per lane per k-step).
// D layout: VGPR r -> M = r + (lane>=16 ? 8 : 0), N = lane&15.
// ---------------------------------------------------------------------------
template <int CT>
__global__ void wmma_gemm_k128(const float* __restrict__ A,
                               const _Float16* __restrict__ Bsw,
                               float* __restrict__ C, int M, int Ncols) {
  const int lane = threadIdx.x & 31;
  const int wave = (int)((blockIdx.x * blockDim.x + threadIdx.x) >> 5);
  const int cgroups = (Ncols >> 4) / CT;
  const int rt = wave / cgroups;
  const int cg = wave - rt * cgroups;
  if (rt * 16 >= M) return;                 // wave-uniform: EXEC stays all-1s
  const int half16 = lane >> 4;             // 0 or 1
  const int l15 = lane & 15;
  int row = rt * 16 + l15;
  if (row >= M) row = M - 1;                // harmless clamp
  const int ga = half16 * 8;                // A K-pattern offset
  const int col0 = cg * CT * 16 + l15;

  v8f acc[CT];
  const v8f vzero = {};
#pragma unroll
  for (int q = 0; q < CT; ++q) acc[q] = vzero;

  for (int k0 = 0; k0 < 128; k0 += 32) {
    v16h a;
    const float* ap = A + (size_t)row * 128 + (k0 + ga);
#pragma unroll
    for (int j = 0; j < 8; ++j) {
      a[j]     = (_Float16)ap[j];           // K = k0+ga+0..7
      a[j + 8] = (_Float16)ap[j + 16];      // K = k0+ga+16..23
    }
    const _Float16* bp =
        Bsw + ((size_t)((k0 >> 5) * 2 + half16) * Ncols + col0) * 16;
#pragma unroll
    for (int q = 0; q < CT; ++q) {
      const v16h b = *(const v16h*)(bp + q * 256);   // next col tile: +16 cols
      acc[q] = __builtin_amdgcn_wmma_f32_16x16x32_f16(false, a, false, b,
                                                      (short)0, acc[q],
                                                      false, false);
    }
  }
  const int mbase = rt * 16 + half16 * 8;
#pragma unroll
  for (int q = 0; q < CT; ++q) {
#pragma unroll
    for (int r = 0; r < 8; ++r) {
      const int m = mbase + r;
      if (m < M) C[(size_t)m * Ncols + (col0 + q * 16)] = acc[q][r];
    }
  }
}

// ---------------------------------------------------------------------------
// v1[k,h] = sum_c We1[k, h*16+c] * att_e1[h,c]   (k<16, h<8)
// v2[k]   = sum_c We2[k, c]      * att_e2[c]     (k<16)
// ---------------------------------------------------------------------------
__global__ void prep_v_kernel(const float* __restrict__ We1,
                              const float* __restrict__ atte1,
                              const float* __restrict__ We2,
                              const float* __restrict__ atte2,
                              float* __restrict__ v1, float* __restrict__ v2) {
  const int t = threadIdx.x;
  if (t < 128) {
    const int k = t >> 3, h = t & 7;
    float s = 0.0f;
    for (int c = 0; c < 16; ++c) s += We1[k * 128 + h * 16 + c] * atte1[h * 16 + c];
    v1[k * 8 + h] = s;
  } else if (t < 144) {
    const int k = t - 128;
    float s = 0.0f;
    for (int c = 0; c < 16; ++c) s += We2[k * 16 + c] * atte2[c];
    v2[k] = s;
  }
}

// a_e1[e,h] = edge_attr[e,:] @ v1[:,h] ;  a_e2[e] = edge_attr[e,:] @ v2
__global__ void edge_ae_kernel(const float* __restrict__ EA,
                               const float* __restrict__ v1,
                               const float* __restrict__ v2,
                               float* __restrict__ ae1, float* __restrict__ ae2,
                               int E) {
  __shared__ float sv1[128];
  __shared__ float sv2[16];
  if (threadIdx.x < 128) sv1[threadIdx.x] = v1[threadIdx.x];
  if (threadIdx.x < 16)  sv2[threadIdx.x] = v2[threadIdx.x];
  __syncthreads();
  const size_t e = (size_t)blockIdx.x * blockDim.x + threadIdx.x;
  if (e >= (size_t)E) return;
  // stream-ahead prefetch of edge_attr (pure HBM stream, ~100 MB)
  __builtin_prefetch(EA + e * 16 + 4096, 0, 1);
  float ea[16];
#pragma unroll
  for (int k = 0; k < 16; ++k) ea[k] = EA[e * 16 + k];
#pragma unroll
  for (int h = 0; h < 8; ++h) {
    float s = 0.0f;
#pragma unroll
    for (int k = 0; k < 16; ++k) s += ea[k] * sv1[k * 8 + h];
    ae1[e * 8 + h] = s;
  }
  float s2 = 0.0f;
#pragma unroll
  for (int k = 0; k < 16; ++k) s2 += ea[k] * sv2[k];
  ae2[e] = s2;
}

// a_s[n,h] = sum_c xh[n,h,c]*att_s[h,c] ; same for a_d
__global__ void node_attn_kernel(const float* __restrict__ xh,
                                 const float* __restrict__ att_s,
                                 const float* __restrict__ att_d,
                                 float* __restrict__ as_, float* __restrict__ ad_,
                                 int Nn, int H, int C) {
  const size_t t = (size_t)blockIdx.x * blockDim.x + threadIdx.x;
  if (t >= (size_t)Nn * H) return;
  const int h = (int)(t % H);
  const float* p = xh + t * C;       // xh[n, h*C .. ] since t = n*H + h
  float s = 0.0f, d = 0.0f;
  for (int c = 0; c < C; ++c) {
    s += p[c] * att_s[h * C + c];
    d += p[c] * att_d[h * C + c];
  }
  as_[t] = s;
  ad_[t] = d;
}

__global__ void fill_kernel(float* __restrict__ p, float v, size_t n) {
  const size_t t = (size_t)blockIdx.x * blockDim.x + threadIdx.x;
  if (t < n) p[t] = v;
}

// Pass 1: logit = leaky_relu(a_s[src]+a_d[dst]+a_e); alpha <- logit; seg max.
__global__ void edge_logit_max_kernel(const int* __restrict__ src,
                                      const int* __restrict__ dst,
                                      const float* __restrict__ as_,
                                      const float* __restrict__ ad_,
                                      float* __restrict__ alpha,
                                      float* __restrict__ m, int E, int H) {
  const size_t t = (size_t)blockIdx.x * blockDim.x + threadIdx.x;
  if (t >= (size_t)E * H) return;
  const size_t e = t / H;
  const int h = (int)(t - e * H);
  const int si = src[e], di = dst[e];
  float l = as_[(size_t)si * H + h] + ad_[(size_t)di * H + h] + alpha[t];
  l = (l > 0.0f) ? l : 0.2f * l;
  alpha[t] = l;
  atomicMaxF(&m[(size_t)di * H + h], l);
}

// Pass 2: alpha <- exp(logit - m[dst]); seg sum.
__global__ void edge_exp_sum_kernel(const int* __restrict__ dst,
                                    float* __restrict__ alpha,
                                    const float* __restrict__ m,
                                    float* __restrict__ s, int E, int H) {
  const size_t t = (size_t)blockIdx.x * blockDim.x + threadIdx.x;
  if (t >= (size_t)E * H) return;
  const size_t e = t / H;
  const int h = (int)(t - e * H);
  const int di = dst[e];
  const float ex = expf(alpha[t] - m[(size_t)di * H + h]);
  alpha[t] = ex;
  atomicAdd(&s[(size_t)di * H + h], ex);
}

// Pass 3: out[dst, f] += xh[src, f] * alpha[e,h]/(s[dst,h]+1e-16), f = h*C+c
__global__ void edge_scatter_kernel(const int* __restrict__ src,
                                    const int* __restrict__ dst,
                                    const float* __restrict__ xh,
                                    const float* __restrict__ alpha,
                                    const float* __restrict__ s,
                                    float* __restrict__ out,
                                    int E, int H, int C) {
  const int F = H * C;
  const size_t t = (size_t)blockIdx.x * blockDim.x + threadIdx.x;
  if (t >= (size_t)E * F) return;
  const size_t e = t / F;
  const int f = (int)(t - e * F);
  const int h = f / C;
  const int si = src[e], di = dst[e];
  const float w = alpha[e * H + h] / (s[(size_t)di * H + h] + 1e-16f);
  atomicAdd(&out[(size_t)di * F + f], xh[(size_t)si * F + f] * w);
}

__global__ void bias_elu_kernel(float* __restrict__ h, const float* __restrict__ b,
                                int Nn, int F) {
  const size_t t = (size_t)blockIdx.x * blockDim.x + threadIdx.x;
  if (t >= (size_t)Nn * F) return;
  const int f = (int)(t % F);
  const float v = h[t] + b[f];
  h[t] = (v > 0.0f) ? v : (expf(v) - 1.0f);
}

// out[n,:] = log_softmax(out[n,:] + b2)
__global__ void finalize_kernel(float* __restrict__ out, const float* __restrict__ b2,
                                int Nn) {
  const size_t n = (size_t)blockIdx.x * blockDim.x + threadIdx.x;
  if (n >= (size_t)Nn) return;
  float v[16];
  float mx = -INFINITY;
#pragma unroll
  for (int c = 0; c < 16; ++c) {
    v[c] = out[n * 16 + c] + b2[c];
    mx = fmaxf(mx, v[c]);
  }
  float sum = 0.0f;
#pragma unroll
  for (int c = 0; c < 16; ++c) sum += expf(v[c] - mx);
  const float l = mx + logf(sum);
#pragma unroll
  for (int c = 0; c < 16; ++c) out[n * 16 + c] = v[c] - l;
}

// ---------------------------------------------------------------------------
extern "C" void kernel_launch(void* const* d_in, const int* in_sizes, int n_in,
                              void* d_out, int out_size, void* d_ws, size_t ws_size,
                              hipStream_t stream) {
  const float* x     = (const float*)d_in[0];
  const int*   ei    = (const int*)d_in[1];
  const float* EA    = (const float*)d_in[2];
  const float* W1    = (const float*)d_in[3];
  const float* atts1 = (const float*)d_in[4];
  const float* attd1 = (const float*)d_in[5];
  const float* We1   = (const float*)d_in[6];
  const float* atte1 = (const float*)d_in[7];
  const float* b1    = (const float*)d_in[8];
  const float* W2    = (const float*)d_in[9];
  const float* atts2 = (const float*)d_in[10];
  const float* attd2 = (const float*)d_in[11];
  const float* We2   = (const float*)d_in[12];
  const float* atte2 = (const float*)d_in[13];
  const float* b2    = (const float*)d_in[14];

  const int N = in_sizes[0] / 128;     // 100000
  const int E = in_sizes[2] / 16;      // 1600000
  const int* src = ei;
  const int* dst = ei + E;

  // Workspace layout (floats; every block is a multiple of 16 floats -> 64B
  // alignment is preserved for the f16 swizzled-weight buffers at the end)
  float* w = (float*)d_ws;
  float* xh1 = w;            w += (size_t)N * 128;
  float* h1  = w;            w += (size_t)N * 128;   // layer-1 accum / ELU output
  float* ae1 = w;            w += (size_t)E * 8;     // a_e1 -> logits -> exp
  float* as1 = w;            w += (size_t)N * 8;
  float* ad1 = w;            w += (size_t)N * 8;
  float* m1  = w;            w += (size_t)N * 8;
  float* s1  = w;            w += (size_t)N * 8;
  float* xh2 = w;            w += (size_t)N * 16;
  float* ae2 = w;            w += (size_t)E;         // a_e2 -> logits -> exp
  float* as2 = w;            w += (size_t)N;
  float* ad2 = w;            w += (size_t)N;
  float* m2  = w;            w += (size_t)N;
  float* s2  = w;            w += (size_t)N;
  float* v1  = w;            w += 128;
  float* v2  = w;            w += 16;
  _Float16* w1sw = (_Float16*)w;                     // 128*128 halfs (32 KB)
  _Float16* w2sw = (_Float16*)(w + 8192);            // 128*16 halfs  (4 KB)
  float* out2 = (float*)d_out;                       // N x 16 accumulator

  const int B = 256;
  #define GRID(total) ((unsigned)(((size_t)(total) + B - 1) / B))
  const float NEG_INF = -INFINITY;

  // --- weight prep: f16 swizzle for WMMA B-operand; edge projections ---
  swizzle_w_f16<<<GRID(128 * 128), B, 0, stream>>>(W1, w1sw, 128, 128);
  swizzle_w_f16<<<GRID(128 * 16), B, 0, stream>>>(W2, w2sw, 128, 16);
  prep_v_kernel<<<1, 160, 0, stream>>>(We1, atte1, We2, atte2, v1, v2);
  edge_ae_kernel<<<GRID(E), B, 0, stream>>>(EA, v1, v2, ae1, ae2, E);

  // ======================== Layer 1 (H=8, C=16) ============================
  {
    const size_t rowTiles = ((size_t)N + 15) / 16;
    const size_t waves = rowTiles * (128 / 16 / 2);   // CT=2 col tiles / wave
    wmma_gemm_k128<2><<<GRID(waves * 32), B, 0, stream>>>(x, w1sw, xh1, N, 128);
  }
  node_attn_kernel<<<GRID((size_t)N * 8), B, 0, stream>>>(xh1, atts1, attd1,
                                                          as1, ad1, N, 8, 16);
  fill_kernel<<<GRID((size_t)N * 8), B, 0, stream>>>(m1, NEG_INF, (size_t)N * 8);
  fill_kernel<<<GRID((size_t)N * 8), B, 0, stream>>>(s1, 0.0f, (size_t)N * 8);
  fill_kernel<<<GRID((size_t)N * 128), B, 0, stream>>>(h1, 0.0f, (size_t)N * 128);

  edge_logit_max_kernel<<<GRID((size_t)E * 8), B, 0, stream>>>(src, dst, as1, ad1,
                                                               ae1, m1, E, 8);
  edge_exp_sum_kernel<<<GRID((size_t)E * 8), B, 0, stream>>>(dst, ae1, m1, s1, E, 8);
  edge_scatter_kernel<<<GRID((size_t)E * 128), B, 0, stream>>>(src, dst, xh1, ae1,
                                                               s1, h1, E, 8, 16);
  bias_elu_kernel<<<GRID((size_t)N * 128), B, 0, stream>>>(h1, b1, N, 128);

  // ======================== Layer 2 (H=1, C=16) ============================
  {
    const size_t rowTiles = ((size_t)N + 15) / 16;
    wmma_gemm_k128<1><<<GRID(rowTiles * 32), B, 0, stream>>>(h1, w2sw, xh2, N, 16);
  }
  node_attn_kernel<<<GRID((size_t)N), B, 0, stream>>>(xh2, atts2, attd2,
                                                      as2, ad2, N, 1, 16);
  fill_kernel<<<GRID((size_t)N), B, 0, stream>>>(m2, NEG_INF, (size_t)N);
  fill_kernel<<<GRID((size_t)N), B, 0, stream>>>(s2, 0.0f, (size_t)N);
  fill_kernel<<<GRID((size_t)N * 16), B, 0, stream>>>(out2, 0.0f, (size_t)N * 16);

  edge_logit_max_kernel<<<GRID((size_t)E), B, 0, stream>>>(src, dst, as2, ad2,
                                                           ae2, m2, E, 1);
  edge_exp_sum_kernel<<<GRID((size_t)E), B, 0, stream>>>(dst, ae2, m2, s2, E, 1);
  edge_scatter_kernel<<<GRID((size_t)E * 16), B, 0, stream>>>(src, dst, xh2, ae2,
                                                              s2, out2, E, 1, 16);
  finalize_kernel<<<GRID((size_t)N), B, 0, stream>>>(out2, b2, N);
  #undef GRID
}
